// AutoSegDecoder_29368986370640
// MI455X (gfx1250) — compile-verified, matrix operand order checked by reading, not run on hardware
//
#include <hip/hip_runtime.h>
#include <hip/hip_bf16.h>

// ---------------------------------------------------------------------------
// Problem constants (B,SQ,SKV,D,H,FF,NL,NSEG,IMG = 4,256,512,512,8,2048,6,32,224)
// ---------------------------------------------------------------------------
#define CB    4
#define CSQ   256
#define CSKV  512
#define CD    512
#define CH    8
#define CDH   64
#define CFF   2048
#define CNL   6
#define CNSEG 32
#define CIMG  224
#define MASKN ((size_t)CB * CNSEG * CIMG * CIMG)   // 6,422,528

typedef __attribute__((ext_vector_type(16))) _Float16 v16h;
typedef __attribute__((ext_vector_type(8)))  float    v8f;

union H16 { _Float16 h[16]; v16h v; };
union F8  { float f[8];     v8f  v; };

// Unguarded: 8 consecutive f32 -> f16 (two global_load_b128 + v_cvt_pk_f16_f32)
__device__ __forceinline__ void cvt8(const float* __restrict__ p, _Float16* dst) {
  const float4* q = reinterpret_cast<const float4*>(p);
  const float4 a = q[0], b = q[1];
  dst[0] = (_Float16)a.x; dst[1] = (_Float16)a.y;
  dst[2] = (_Float16)a.z; dst[3] = (_Float16)a.w;
  dst[4] = (_Float16)b.x; dst[5] = (_Float16)b.y;
  dst[6] = (_Float16)b.z; dst[7] = (_Float16)b.w;
}

// ---------------------------------------------------------------------------
// WMMA GEMM:  C = act(A @ W^T + bias),  W is [N,K] row-major.
// f32 operands, f16 in-register conversion, f32 accumulation
// (v_wmma_f32_16x16x32_f16).  Batched over blockIdx.z = b*Hn + h.
// Block = 256 threads (8 waves); wave -> 16(M) x 64(N); block -> 128 x 64.
// Out-of-range A/W rows are CLAMPED (their products only land in never-stored
// output rows/cols), so the main K-loop is completely branch-free; only the
// K%32 tail (K=5096) takes a single guarded epilogue step.
// ---------------------------------------------------------------------------
template <int ACT>
__global__ __launch_bounds__(256) void gemm_kernel(
    const float* __restrict__ A, int lda, size_t sAb, size_t sAh,
    const float* __restrict__ W, int ldw, size_t sWb, size_t sWh,
    const float* __restrict__ bias,
    float* __restrict__ C, int ldc, size_t sCb, size_t sCh,
    int M, int N, int K, int Hn)
{
  const int z  = blockIdx.z;
  const int zb = z / Hn, zh = z % Hn;
  A += (size_t)zb * sAb + (size_t)zh * sAh;
  W += (size_t)zb * sWb + (size_t)zh * sWh;
  C += (size_t)zb * sCb + (size_t)zh * sCh;

  const int wave  = threadIdx.x >> 5;
  const int lane  = threadIdx.x & 31;
  const int lrow  = lane & 15;
  const int lhalf = lane >> 4;                 // 0 or 1

  const int m0 = blockIdx.y * 128 + wave * 16;
  const int n0 = blockIdx.x * 64;

  v8f acc[4];
  {
    F8 zf;
#pragma unroll
    for (int r = 0; r < 8; ++r) zf.f[r] = 0.0f;
#pragma unroll
    for (int j = 0; j < 4; ++j) acc[j] = zf.v;
  }

  const int am = m0 + lrow;
  const float* Arow = A + (size_t)(am < M ? am : M - 1) * lda;

  const float* Wr[4];
#pragma unroll
  for (int j = 0; j < 4; ++j) {
    const int n = n0 + j * 16 + lrow;
    Wr[j] = W + (size_t)(n < N ? n : N - 1) * ldw;
  }

  const int Kmain = K & ~31;
  for (int kb = 0; kb < Kmain; kb += 32) {
    const int kb0 = kb + lhalf * 8;            // ISA 16-bit A/B per-lane K base
    H16 a;
    cvt8(Arow + kb0,      a.h);
    cvt8(Arow + kb0 + 16, a.h + 8);
#pragma unroll
    for (int j = 0; j < 4; ++j) {
      H16 w;
      cvt8(Wr[j] + kb0,      w.h);
      cvt8(Wr[j] + kb0 + 16, w.h + 8);
      acc[j] = __builtin_amdgcn_wmma_f32_16x16x32_f16(
          false, a.v, false, w.v, (short)0, acc[j], false, false);
    }
  }

  if (Kmain < K) {                             // single guarded tail step
    const int kb0 = Kmain + lhalf * 8;
    H16 a;
#pragma unroll
    for (int i = 0; i < 8; ++i) {
      const int k0 = kb0 + i, k1 = kb0 + 16 + i;
      a.h[i]     = (k0 < K) ? (_Float16)Arow[k0] : (_Float16)0.0f;
      a.h[8 + i] = (k1 < K) ? (_Float16)Arow[k1] : (_Float16)0.0f;
    }
#pragma unroll
    for (int j = 0; j < 4; ++j) {
      H16 w;
#pragma unroll
      for (int i = 0; i < 8; ++i) {
        const int k0 = kb0 + i, k1 = kb0 + 16 + i;
        w.h[i]     = (k0 < K) ? (_Float16)Wr[j][k0] : (_Float16)0.0f;
        w.h[8 + i] = (k1 < K) ? (_Float16)Wr[j][k1] : (_Float16)0.0f;
      }
      acc[j] = __builtin_amdgcn_wmma_f32_16x16x32_f16(
          false, a.v, false, w.v, (short)0, acc[j], false, false);
    }
  }

#pragma unroll
  for (int j = 0; j < 4; ++j) {
    const int n = n0 + j * 16 + lrow;
    if (n >= N) continue;
    const float bv = bias ? bias[n] : 0.0f;
    F8 u; u.v = acc[j];
#pragma unroll
    for (int r = 0; r < 8; ++r) {
      const int m = m0 + lhalf * 8 + r;        // C layout: VGPR r + lane-half -> M
      if (m < M) {
        float f = u.f[r] + bv;
        if (ACT == 1)      f = f > 0.0f ? f : 0.0f;        // relu
        else if (ACT == 2) f = f > 0.0f ? f : 0.01f * f;   // leaky_relu
        C[(size_t)m * ldc + n] = f;
      }
    }
  }
}

// ---------------------------------------------------------------------------
// Per-(b,h) transpose of V: vh[B*skv, 512] head-sliced  ->  vt[z*64+d][k]
// LDS 64x64 tile so both global sides are coalesced.
// ---------------------------------------------------------------------------
__global__ __launch_bounds__(256) void transpose_v_kernel(
    const float* __restrict__ vh, float* __restrict__ vt, int skv)
{
  __shared__ float tile[64][65];
  const int z = blockIdx.y, b = z / CH, h = z % CH;
  const int k0 = blockIdx.x * 64;
  const int t = threadIdx.x;
#pragma unroll
  for (int i = 0; i < 16; ++i) {
    const int e = t + i * 256;
    const int r = e >> 6, c = e & 63;          // r: k offset, c: d offset
    tile[r][c] = vh[(size_t)(b * skv + k0 + r) * CD + h * CDH + c];
  }
  __syncthreads();
#pragma unroll
  for (int i = 0; i < 16; ++i) {
    const int e = t + i * 256;
    const int r = e >> 6, c = e & 63;          // r: d, c: k offset
    vt[(size_t)(z * CDH + r) * skv + k0 + c] = tile[c][r];
  }
}

// ---------------------------------------------------------------------------
// Row softmax with optional causal mask; one wave per row.
// ---------------------------------------------------------------------------
__global__ __launch_bounds__(256) void softmax_kernel(
    float* __restrict__ s, int rows, int cols, int sq, int causal, float scale)
{
  const int row  = blockIdx.x * 8 + (threadIdx.x >> 5);
  const int lane = threadIdx.x & 31;
  if (row >= rows) return;
  float* p = s + (size_t)row * cols;
  const int qidx = row % sq;
  const int nit  = cols >> 5;                  // 8 or 16

  float v[16];
  float mx = -3.4e38f;
  for (int i = 0; i < nit; ++i) {
    const int j = lane + i * 32;
    float f = p[j] * scale;
    if (causal && j > qidx) f += -1e9f;
    v[i] = f;
    mx = fmaxf(mx, f);
  }
  for (int off = 16; off; off >>= 1) mx = fmaxf(mx, __shfl_xor(mx, off, 32));
  float sum = 0.0f;
  for (int i = 0; i < nit; ++i) { v[i] = __expf(v[i] - mx); sum += v[i]; }
  for (int off = 16; off; off >>= 1) sum += __shfl_xor(sum, off, 32);
  const float inv = 1.0f / sum;
  for (int i = 0; i < nit; ++i) p[lane + i * 32] = v[i] * inv;
}

// ---------------------------------------------------------------------------
// x = LayerNorm(x + r) * g + b   (row = 512, one wave per row, in place)
// ---------------------------------------------------------------------------
__global__ __launch_bounds__(256) void add_ln_kernel(
    float* __restrict__ x, const float* __restrict__ r,
    const float* __restrict__ g, const float* __restrict__ b, int rows)
{
  const int row  = blockIdx.x * 8 + (threadIdx.x >> 5);
  const int lane = threadIdx.x & 31;
  if (row >= rows) return;
  float* xr = x + (size_t)row * CD;
  const float* rr = r + (size_t)row * CD;

  float v[16];
  float s = 0.0f;
#pragma unroll
  for (int i = 0; i < 16; ++i) { const int j = lane + i * 32; v[i] = xr[j] + rr[j]; s += v[i]; }
  for (int off = 16; off; off >>= 1) s += __shfl_xor(s, off, 32);
  const float mean = s * (1.0f / CD);
  float q = 0.0f;
#pragma unroll
  for (int i = 0; i < 16; ++i) { const float d = v[i] - mean; q += d * d; }
  for (int off = 16; off; off >>= 1) q += __shfl_xor(q, off, 32);
  const float inv = rsqrtf(q * (1.0f / CD) + 1e-5f);
#pragma unroll
  for (int i = 0; i < 16; ++i) {
    const int j = lane + i * 32;
    xr[j] = (v[i] - mean) * inv * g[j] + b[j];
  }
}

// ---------------------------------------------------------------------------
// Consecutive-run segment sum: pooled[b, seg, :] = sum over run of h[b, s, :]
// ---------------------------------------------------------------------------
__global__ __launch_bounds__(256) void segsum_kernel(
    const float* __restrict__ h, const int* __restrict__ labels,
    float* __restrict__ pooled)
{
  const int b = blockIdx.x;
  const int t = threadIdx.x;
  __shared__ int seg[CSQ];

  float* pb = pooled + (size_t)b * CNSEG * CD;
  for (int i = t; i < CNSEG * CD; i += 256) pb[i] = 0.0f;

  if (t == 0) {
    int s = 0, prev = labels[b * CSQ];
    seg[0] = 0;
    for (int i = 1; i < CSQ; ++i) {
      const int cur = labels[b * CSQ + i];
      if (cur != prev) ++s;
      prev = cur;
      seg[i] = s;
    }
  }
  __syncthreads();

  const float* hb = h + (size_t)b * CSQ * CD;
  for (int d = t; d < CD; d += 256) {
    float acc = 0.0f;
    int cs = seg[0];
    for (int s = 0; s < CSQ; ++s) {
      const int sg = seg[s];
      if (sg != cs) {
        if (cs < CNSEG) pb[(size_t)cs * CD + d] = acc;
        acc = 0.0f; cs = sg;
      }
      acc += hb[(size_t)s * CD + d];
    }
    if (cs < CNSEG) pb[(size_t)cs * CD + d] = acc;
  }
}

// ---------------------------------------------------------------------------
// Host-side dispatch
// ---------------------------------------------------------------------------
static inline void launch_gemm(hipStream_t st,
    const float* A, int lda, size_t sAb, size_t sAh,
    const float* W, int ldw, size_t sWb, size_t sWh,
    const float* bias,
    float* C, int ldc, size_t sCb, size_t sCh,
    int M, int N, int K, int batches, int Hn, int act)
{
  dim3 grid((unsigned)((N + 63) / 64), (unsigned)((M + 127) / 128), (unsigned)batches);
  dim3 block(256);
#define GL(ACT) gemm_kernel<ACT><<<grid, block, 0, st>>>( \
      A, lda, sAb, sAh, W, ldw, sWb, sWh, bias, C, ldc, sCb, sCh, M, N, K, Hn)
  if (act == 0)      GL(0);
  else if (act == 1) GL(1);
  else               GL(2);
#undef GL
}

// Workspace layout (f32 elements). Scores alias the MA ping buffer (temporally
// disjoint: scores live only inside the decoder loop, MA only in the heads).
#define OX   ((size_t)0)                       // x           1024*512
#define OQB  ((size_t)524288)                  // q proj      1024*512
#define OKB  ((size_t)1048576)                 // k proj      2048*512
#define OVB  ((size_t)2097152)                 // v proj      2048*512
#define OCTX ((size_t)3145728)                 // attn ctx    1024*512
#define OFF_ ((size_t)3670016)                 // ffn hidden  1024*2048
#define OMA  ((size_t)5767168)                 // mlp ping    1024*5096 (alias: scores 4*8*256*512)
#define OMB  ((size_t)10985472)                // mlp pong    1024*5096
#define OPL  ((size_t)16203776)                // pooled      128*512
#define OVT  ((size_t)16269312)                // V^T         32*64*512

extern "C" void kernel_launch(void* const* d_in, const int* in_sizes, int n_in,
                              void* d_out, int out_size, void* d_ws, size_t ws_size,
                              hipStream_t stream)
{
  (void)in_sizes; (void)out_size; (void)ws_size;

  const float* vk = (const float*)d_in[0];       // [B,SKV,D]
  const float* qs = (const float*)d_in[1];       // [B,SQ,D]
  const int*  tsi = (const int*)d_in[n_in - 2];  // [B,SQ]
  const int pb = 2;                              // params leaves start here
  auto P = [&](int i) { return (const float*)d_in[i]; };

  float* ws   = (float*)d_ws;
  float* X    = ws + OX;
  float* QB   = ws + OQB;
  float* KB   = ws + OKB;
  float* VB   = ws + OVB;
  float* CTX  = ws + OCTX;
  float* FFB  = ws + OFF_;
  float* MA   = ws + OMA;
  float* MB   = ws + OMB;
  float* POOL = ws + OPL;
  float* VT   = ws + OVT;
  float* ATT  = MA;                              // aliased scores buffer

  const int MT  = CB * CSQ;                      // 1024 token rows
  const int MKV = CB * CSKV;                     // 2048 kv rows
  const float scale = 0.125f;                    // 1/sqrt(DH)

  hipMemcpyAsync(X, qs, sizeof(float) * (size_t)MT * CD, hipMemcpyDeviceToDevice, stream);

  for (int l = 0; l < CNL; ++l) {
    const int lb = pb + 8 + 26 * l;
    // leaf order within layer (sorted keys):
    // ca: k.b k.w o.b o.w q.b q.w v.b v.w | ff1.b ff1.w | ff2.b ff2.w |
    // ln1.b ln1.g | ln2.b ln2.g | ln3.b ln3.g | sa: k.b k.w o.b o.w q.b q.w v.b v.w

    // ---------------- self-attention (causal), skv = SQ ----------------
    launch_gemm(stream, X, CD, 0, 0, P(lb+23), CD, 0, 0, P(lb+22), QB, CD, 0, 0, MT, CD, CD, 1, 1, 0);
    launch_gemm(stream, X, CD, 0, 0, P(lb+19), CD, 0, 0, P(lb+18), KB, CD, 0, 0, MT, CD, CD, 1, 1, 0);
    launch_gemm(stream, X, CD, 0, 0, P(lb+25), CD, 0, 0, P(lb+24), VB, CD, 0, 0, MT, CD, CD, 1, 1, 0);
    transpose_v_kernel<<<dim3(CSQ / 64, CB * CH), dim3(256), 0, stream>>>(VB, VT, CSQ);
    launch_gemm(stream, QB, CD, (size_t)CSQ * CD, CDH,
                        KB, CD, (size_t)CSQ * CD, CDH, nullptr,
                        ATT, CSQ, (size_t)CH * CSQ * CSQ, (size_t)CSQ * CSQ,
                        CSQ, CSQ, CDH, CB * CH, CH, 0);
    softmax_kernel<<<dim3((CB * CH * CSQ + 7) / 8), dim3(256), 0, stream>>>(
        ATT, CB * CH * CSQ, CSQ, CSQ, 1, scale);
    launch_gemm(stream, ATT, CSQ, (size_t)CH * CSQ * CSQ, (size_t)CSQ * CSQ,
                        VT, CSQ, (size_t)CH * CDH * CSQ, (size_t)CDH * CSQ, nullptr,
                        CTX, CD, (size_t)CSQ * CD, CDH,
                        CSQ, CDH, CSQ, CB * CH, CH, 0);
    launch_gemm(stream, CTX, CD, 0, 0, P(lb+21), CD, 0, 0, P(lb+20), QB, CD, 0, 0, MT, CD, CD, 1, 1, 0);
    add_ln_kernel<<<dim3(MT / 8), dim3(256), 0, stream>>>(X, QB, P(lb+13), P(lb+12), MT);

    // ---------------- cross-attention, skv = SKV ----------------
    launch_gemm(stream, X,  CD, 0, 0, P(lb+5), CD, 0, 0, P(lb+4), QB, CD, 0, 0, MT,  CD, CD, 1, 1, 0);
    launch_gemm(stream, vk, CD, 0, 0, P(lb+1), CD, 0, 0, P(lb+0), KB, CD, 0, 0, MKV, CD, CD, 1, 1, 0);
    launch_gemm(stream, vk, CD, 0, 0, P(lb+7), CD, 0, 0, P(lb+6), VB, CD, 0, 0, MKV, CD, CD, 1, 1, 0);
    transpose_v_kernel<<<dim3(CSKV / 64, CB * CH), dim3(256), 0, stream>>>(VB, VT, CSKV);
    launch_gemm(stream, QB, CD, (size_t)CSQ * CD, CDH,
                        KB, CD, (size_t)CSKV * CD, CDH, nullptr,
                        ATT, CSKV, (size_t)CH * CSQ * CSKV, (size_t)CSQ * CSKV,
                        CSQ, CSKV, CDH, CB * CH, CH, 0);
    softmax_kernel<<<dim3((CB * CH * CSQ + 7) / 8), dim3(256), 0, stream>>>(
        ATT, CB * CH * CSQ, CSKV, CSQ, 0, scale);
    launch_gemm(stream, ATT, CSKV, (size_t)CH * CSQ * CSKV, (size_t)CSQ * CSKV,
                        VT, CSKV, (size_t)CH * CDH * CSKV, (size_t)CDH * CSKV, nullptr,
                        CTX, CD, (size_t)CSQ * CD, CDH,
                        CSQ, CDH, CSKV, CB * CH, CH, 0);
    launch_gemm(stream, CTX, CD, 0, 0, P(lb+3), CD, 0, 0, P(lb+2), QB, CD, 0, 0, MT, CD, CD, 1, 1, 0);
    add_ln_kernel<<<dim3(MT / 8), dim3(256), 0, stream>>>(X, QB, P(lb+15), P(lb+14), MT);

    // ---------------- FFN ----------------
    launch_gemm(stream, X,   CD,  0, 0, P(lb+9),  CD,  0, 0, P(lb+8),  FFB, CFF, 0, 0, MT, CFF, CD,  1, 1, 1);
    launch_gemm(stream, FFB, CFF, 0, 0, P(lb+11), CFF, 0, 0, P(lb+10), QB,  CD,  0, 0, MT, CD,  CFF, 1, 1, 0);
    add_ln_kernel<<<dim3(MT / 8), dim3(256), 0, stream>>>(X, QB, P(lb+17), P(lb+16), MT);
  }

  // ---------------- label head: 512 -> 1024 -> 2048 -> 5096 -> 512 ----------------
  float* outLabel = (float*)d_out + MASKN;
  launch_gemm(stream, X,  CD,   0,0, P(pb+1), CD,   0,0, P(pb+0), MA, 1024, 0,0, MT, 1024, CD,   1,1, 2);
  launch_gemm(stream, MA, 1024, 0,0, P(pb+3), 1024, 0,0, P(pb+2), MB, 2048, 0,0, MT, 2048, 1024, 1,1, 2);
  launch_gemm(stream, MB, 2048, 0,0, P(pb+5), 2048, 0,0, P(pb+4), MA, 5096, 0,0, MT, 5096, 2048, 1,1, 2);
  launch_gemm(stream, MA, 5096, 0,0, P(pb+7), 5096, 0,0, P(pb+6), outLabel, CD, 0,0, MT, CD, 5096, 1,1, 0);

  // ---------------- segment-sum pooling ----------------
  segsum_kernel<<<dim3(CB), dim3(256), 0, stream>>>(X, tsi, POOL);

  // ---------------- pixel head: 512 -> 1024 -> 2048 -> 5096 -> 50176 ----------------
  const int px = pb + 8 + 26 * CNL;
  const int MP = CB * CNSEG;                     // 128 pooled rows
  float* outMask = (float*)d_out;
  launch_gemm(stream, POOL, CD,  0,0, P(px+1), CD,   0,0, P(px+0), MA, 1024, 0,0, MP, 1024, CD,   1,1, 2);
  launch_gemm(stream, MA, 1024,  0,0, P(px+3), 1024, 0,0, P(px+2), MB, 2048, 0,0, MP, 2048, 1024, 1,1, 2);
  launch_gemm(stream, MB, 2048,  0,0, P(px+5), 2048, 0,0, P(px+4), MA, 5096, 0,0, MP, 5096, 2048, 1,1, 2);
  launch_gemm(stream, MA, 5096,  0,0, P(px+7), 5096, 0,0, P(px+6), outMask, CIMG * CIMG, 0,0,
              MP, CIMG * CIMG, 5096, 1,1, 0);
}
